// ScaledDotProductAttention_31516470018113
// MI455X (gfx1250) — matrix operand order repaired and checked by reading, not compiled, not run on hardware
//
#include <hip/hip_runtime.h>

typedef __attribute__((ext_vector_type(16))) __bf16 v16bf;
typedef __attribute__((ext_vector_type(8)))  float  v8f;

#define BB 8
#define SS 2048
#define DD 128
#define BM 128            // query rows per block
#define BN 32             // kv rows per tile
#define NWAVE 8
#define KSTRIDE 136       // ushorts per K LDS row (128 + 8 pad) -> 272B, 16B aligned
#define VSTRIDE 48        // ushorts per V^T LDS row (32 + 16 pad) -> 96B, 16B aligned
#define PSTRIDE 48        // ushorts per P LDS row

__device__ __forceinline__ unsigned short f2bf(float f) {
  unsigned u = __float_as_uint(f);
  u += 0x7FFFu + ((u >> 16) & 1u);      // round-to-nearest-even
  return (unsigned short)(u >> 16);
}

// Split workgroup barrier that only drains DScnt (publishes LDS stores and
// retires this wave's LDS reads of the previous buffer) while leaving global
// LOADcnt in flight — unlike __syncthreads(), whose fence drains LOADcnt too.
__device__ __forceinline__ void wg_barrier_signal_dswait() {
  asm volatile("s_wait_dscnt 0x0\n\ts_barrier_signal -1" ::: "memory");
}
__device__ __forceinline__ void wg_barrier_wait() {
  asm volatile("s_barrier_wait -1" ::: "memory");
}

union FragB { v16bf v; uint4 q[2]; unsigned short s[16]; };

__global__ __launch_bounds__(256)
void fa_fwd_causal(const float* __restrict__ Q, const float* __restrict__ K,
                   const float* __restrict__ V, float* __restrict__ O) {
  // double-buffered K (row-major bf16) and V^T (bf16) staging tiles
  __shared__ unsigned short ksm[2 * BN * KSTRIDE];
  __shared__ unsigned short vsm[2 * DD * VSTRIDE];
  __shared__ unsigned short psm[NWAVE * 16 * PSTRIDE];

  const int tid = threadIdx.x;
  const int w  = tid >> 5;
  const int l  = tid & 31;
  const int lh = l >> 4;        // which 16-lane half
  const int ll = l & 15;

  const int nqb = SS / BM;
  const int b   = blockIdx.x / nqb;
  const int qb  = blockIdx.x % nqb;
  const int q0  = qb * BM;
  const int qrw = q0 + w * 16;  // this wave's first query row

  const float* Qg = Q + (size_t)b * SS * DD;
  const float* Kg = K + (size_t)b * SS * DD;
  const float* Vg = V + (size_t)b * SS * DD;
  float*       Og = O + (size_t)b * SS * DD;

  const float sc = 0.08838834764831845f * 1.4426950408889634f; // 1/sqrt(128) * log2(e)

  // ---- Q fragments (A layout), scaled, bf16, kept in registers ----
  FragB qf[4];
  {
    const float* qp = Qg + (size_t)(qrw + ll) * DD;
    for (int c = 0; c < 4; ++c) {
      const int base = c * 32 + lh * 8;
      float4 a0 = *(const float4*)(qp + base);
      float4 a1 = *(const float4*)(qp + base + 4);
      float4 a2 = *(const float4*)(qp + base + 16);
      float4 a3 = *(const float4*)(qp + base + 20);
      float t[16] = {a0.x,a0.y,a0.z,a0.w, a1.x,a1.y,a1.z,a1.w,
                     a2.x,a2.y,a2.z,a2.w, a3.x,a3.y,a3.z,a3.w};
      for (int j = 0; j < 16; ++j) qf[c].s[j] = f2bf(t[j] * sc);
    }
  }

  const v8f vzero = {};
  v8f acc[8];
  for (int t = 0; t < 8; ++t) acc[t] = vzero;
  float mrow[8], lrow[8];
  for (int r = 0; r < 8; ++r) { mrow[r] = -3.0e38f; lrow[r] = 0.0f; }

  const int nkt = (q0 + BM) / BN;       // causal upper bound for this block

  // ---- software-pipelined staging registers: tile kt held in kreg/vreg ----
  float4 kreg[4], vreg[4];
  for (int i = 0; i < 4; ++i) {
    const int e   = tid + i * 256;
    const int row = e >> 5;
    const int c4  = e & 31;
    kreg[i] = *(const float4*)(Kg + (size_t)row * DD + c4 * 4);
    vreg[i] = *(const float4*)(Vg + (size_t)row * DD + c4 * 4);
  }

  for (int kt = 0; kt < nkt; ++kt) {
    const int kv0 = kt * BN;
    unsigned short* kb = ksm + (kt & 1) * (BN * KSTRIDE);
    unsigned short* vb = vsm + (kt & 1) * (DD * VSTRIDE);

    // ---- convert + store tile kt (already in registers) into LDS ----
    for (int i = 0; i < 4; ++i) {
      const int e   = tid + i * 256;
      const int row = e >> 5;
      const int c4  = e & 31;
      unsigned lo = (unsigned)f2bf(kreg[i].x) | ((unsigned)f2bf(kreg[i].y) << 16);
      unsigned hi = (unsigned)f2bf(kreg[i].z) | ((unsigned)f2bf(kreg[i].w) << 16);
      *(uint2*)&kb[row * KSTRIDE + c4 * 4] = make_uint2(lo, hi);
      vb[(c4 * 4 + 0) * VSTRIDE + row] = f2bf(vreg[i].x);
      vb[(c4 * 4 + 1) * VSTRIDE + row] = f2bf(vreg[i].y);
      vb[(c4 * 4 + 2) * VSTRIDE + row] = f2bf(vreg[i].z);
      vb[(c4 * 4 + 3) * VSTRIDE + row] = f2bf(vreg[i].w);
    }

    // ---- split barrier: publish LDS stores (DScnt only), then issue the
    //      next tile's global loads between signal and wait so they remain
    //      in flight across the barrier and the whole compute phase ----
    wg_barrier_signal_dswait();

    if (kt + 1 < nkt) {
      const int kv1 = kv0 + BN;
      for (int i = 0; i < 4; ++i) {
        const int e   = tid + i * 256;
        const int row = e >> 5;
        const int c4  = e & 31;
        kreg[i] = *(const float4*)(Kg + (size_t)(kv1 + row) * DD + c4 * 4);
        vreg[i] = *(const float4*)(Vg + (size_t)(kv1 + row) * DD + c4 * 4);
      }
    }
    if (kt + 2 < nkt) {                 // L2 prefetch two tiles ahead
      __builtin_prefetch(Kg + (size_t)(kv0 + 2 * BN) * DD + tid * 16, 0, 0);
      __builtin_prefetch(Vg + (size_t)(kv0 + 2 * BN) * DD + tid * 16, 0, 0);
    }

    wg_barrier_wait();

    if (kv0 <= qrw + 15) {              // wave-uniform causal skip (EXEC stays full)
      // ---- S = Q K^T : 8 WMMAs ----
      v8f s0 = vzero, s1 = vzero;
      for (int c = 0; c < 4; ++c) {
        const int base = c * 32 + lh * 8;
        FragB kf0, kf1;
        kf0.q[0] = *(const uint4*)&kb[ll * KSTRIDE + base];
        kf0.q[1] = *(const uint4*)&kb[ll * KSTRIDE + base + 16];
        kf1.q[0] = *(const uint4*)&kb[(16 + ll) * KSTRIDE + base];
        kf1.q[1] = *(const uint4*)&kb[(16 + ll) * KSTRIDE + base + 16];
        s0 = __builtin_amdgcn_wmma_f32_16x16x32_bf16(false, qf[c].v, false, kf0.v,
                                                     (short)0, s0, false, false);
        s1 = __builtin_amdgcn_wmma_f32_16x16x32_bf16(false, qf[c].v, false, kf1.v,
                                                     (short)0, s1, false, false);
      }
      // ---- causal mask on diagonal-crossing tiles ----
      if (kv0 + BN - 1 > qrw) {
        for (int r = 0; r < 8; ++r) {
          const int qr = qrw + r + 8 * lh;
          if (kv0 + ll > qr)      s0[r] = -3.0e38f;
          if (kv0 + 16 + ll > qr) s1[r] = -3.0e38f;
        }
      }
      // ---- online softmax (log2 domain; row stats via half-wave xor shuffles) ----
      float alpha[8];
      for (int r = 0; r < 8; ++r) {
        float ml = fmaxf(s0[r], s1[r]);
        ml = fmaxf(ml, __shfl_xor(ml, 1, 32));
        ml = fmaxf(ml, __shfl_xor(ml, 2, 32));
        ml = fmaxf(ml, __shfl_xor(ml, 4, 32));
        ml = fmaxf(ml, __shfl_xor(ml, 8, 32));
        const float mn = fmaxf(mrow[r], ml);
        alpha[r] = exp2f(mrow[r] - mn);
        mrow[r] = mn;
        const float p0 = exp2f(s0[r] - mn);
        const float p1 = exp2f(s1[r] - mn);
        s0[r] = p0; s1[r] = p1;
        float sl = p0 + p1;
        sl += __shfl_xor(sl, 1, 32);
        sl += __shfl_xor(sl, 2, 32);
        sl += __shfl_xor(sl, 4, 32);
        sl += __shfl_xor(sl, 8, 32);
        lrow[r] = lrow[r] * alpha[r] + sl;
      }
      for (int t = 0; t < 8; ++t)
        for (int r = 0; r < 8; ++r) acc[t][r] *= alpha[r];

      // ---- P: C-layout -> A-fragment via per-wave LDS round trip ----
      unsigned short* ps = &psm[w * 16 * PSTRIDE];
      for (int r = 0; r < 8; ++r) {
        const int row = r + 8 * lh;
        ps[row * PSTRIDE + ll]      = f2bf(s0[r]);
        ps[row * PSTRIDE + 16 + ll] = f2bf(s1[r]);
      }
      FragB pf;
      pf.q[0] = *(const uint4*)&ps[ll * PSTRIDE + lh * 8];
      pf.q[1] = *(const uint4*)&ps[ll * PSTRIDE + lh * 8 + 16];

      // ---- O += P V : 8 WMMAs (B-fragments from transposed V in LDS) ----
      for (int t = 0; t < 8; ++t) {
        const int vr = (t * 16 + ll) * VSTRIDE + lh * 8;
        FragB vf;
        vf.q[0] = *(const uint4*)&vb[vr];
        vf.q[1] = *(const uint4*)&vb[vr + 16];
        acc[t] = __builtin_amdgcn_wmma_f32_16x16x32_bf16(false, pf.v, false, vf.v,
                                                         (short)0, acc[t], false, false);
      }
    }
    // no trailing barrier: next iteration writes the other LDS buffer
  }

  // ---- normalize and write O (fp32) ----
  for (int r = 0; r < 8; ++r) {
    const float inv = 1.0f / lrow[r];
    const int row = qrw + r + 8 * lh;
    for (int t = 0; t < 8; ++t) {
      Og[(size_t)row * DD + t * 16 + ll] = acc[t][r] * inv;
    }
  }
}

extern "C" void kernel_launch(void* const* d_in, const int* in_sizes, int n_in,
                              void* d_out, int out_size, void* d_ws, size_t ws_size,
                              hipStream_t stream) {
  (void)in_sizes; (void)n_in; (void)out_size; (void)d_ws; (void)ws_size;
  const float* Q = (const float*)d_in[0];
  const float* K = (const float*)d_in[1];
  const float* V = (const float*)d_in[2];
  // d_in[3] is the tril mask; causality is computed analytically in-kernel.
  float* O = (float*)d_out;
  dim3 grid(BB * (SS / BM));
  fa_fwd_causal<<<grid, 256, 0, stream>>>(Q, K, V, O);
}
